// MixtralBlockSparseTop2MLP_4398046511748
// MI455X (gfx1250) — matrix operand order repaired and checked by reading, not compiled, not run on hardware
//
#include <hip/hip_runtime.h>
#include <hip/hip_bf16.h>
#include <stdint.h>

typedef __bf16 bf16;
typedef __attribute__((ext_vector_type(16))) __bf16 v16bf;
typedef __attribute__((ext_vector_type(8)))  __bf16 v8bf;
typedef __attribute__((ext_vector_type(8)))  float   v8f;

#define T_TOK 8192
#define DDIM  1024
#define NEXP  8
#define FDIM  3584
#define KTOP  2
#define NASSIGN (T_TOK * KTOP)

union BFrag { v16bf v; v8bf h[2]; };

#define WMMA_BF16(acc, A, B) \
    (acc) = __builtin_amdgcn_wmma_f32_16x16x32_bf16(false, (A).v, false, (B).v, \
                                                    (short)0, (acc), false, false)

// A 16x32 bf16 frag: lanes 0-15 hold K 0..7 & 16..23 of row M=lane; lanes 16-31 hold K 8..15 & 24..31
#define LOAD_AFRAG(dst, base, k, lhi) do {                         \
    (dst).h[0] = *(const v8bf*)((base) + (k) + (lhi) * 8);         \
    (dst).h[1] = *(const v8bf*)((base) + (k) + 16 + (lhi) * 8);    \
} while (0)
// B 32x16 bf16 frag: lane holds one weight row (N); lanes 0-15 hold K 0..7 & 16..23, lanes 16-31 hold K 8..15 & 24..31
#define LOAD_BFRAG(dst, base, k, lhi) do {                         \
    (dst).h[0] = *(const v8bf*)((base) + (k) + (lhi) * 8);         \
    (dst).h[1] = *(const v8bf*)((base) + (k) + 16 + (lhi) * 8);    \
} while (0)

__device__ __forceinline__ unsigned short bf16_bits(float f) {
    union { float f; uint32_t u; } v; v.f = f;
    uint32_t r = v.u + 0x7fffu + ((v.u >> 16) & 1u);   // round-to-nearest-even
    return (unsigned short)(r >> 16);
}
__device__ __forceinline__ bf16 to_bf16(float f) {
    unsigned short hs = bf16_bits(f);
    return __builtin_bit_cast(bf16, hs);
}

// ---------------- fp32 -> bf16 conversion (vectorized, grid-stride) ----------------
__global__ void cvt_f32_bf16_kernel(const float* __restrict__ src,
                                    unsigned short* __restrict__ dst, long long n4) {
    long long i = (long long)blockIdx.x * blockDim.x + threadIdx.x;
    long long stride = (long long)gridDim.x * blockDim.x;
    for (; i < n4; i += stride) {
        float4 f = ((const float4*)src)[i];
        ushort4 o;
        o.x = bf16_bits(f.x); o.y = bf16_bits(f.y);
        o.z = bf16_bits(f.z); o.w = bf16_bits(f.w);
        ((ushort4*)dst)[i] = o;
    }
}

// ---------------- routing: softmax + top2 + counts ----------------
__global__ void route_kernel(const float* __restrict__ logits,
                             int* __restrict__ counts,
                             int* __restrict__ texp, float* __restrict__ twgt) {
    int t = blockIdx.x * blockDim.x + threadIdx.x;
    if (t >= T_TOK) return;
    float l[NEXP], p[NEXP];
    float mx = -1e30f;
    for (int e = 0; e < NEXP; ++e) { l[e] = logits[t * NEXP + e]; mx = fmaxf(mx, l[e]); }
    for (int e = 0; e < NEXP; ++e) { p[e] = __expf(l[e] - mx); }
    int i0 = 0;
    for (int e = 1; e < NEXP; ++e) if (p[e] > p[i0]) i0 = e;        // first-occurrence max
    int i1 = (i0 == 0) ? 1 : 0;
    for (int e = 0; e < NEXP; ++e) if (e != i0 && p[e] > p[i1]) i1 = e;
    float inv = 1.0f / (p[i0] + p[i1]);                              // renormalize top-2
    texp[t * 2 + 0] = i0;  twgt[t * 2 + 0] = p[i0] * inv;
    texp[t * 2 + 1] = i1;  twgt[t * 2 + 1] = p[i1] * inv;
    atomicAdd(&counts[i0], 1);
    atomicAdd(&counts[i1], 1);
}

__global__ void scan_kernel(const int* __restrict__ counts,
                            int* __restrict__ offsets, int* __restrict__ cursor) {
    if (threadIdx.x == 0 && blockIdx.x == 0) {
        int acc = 0;
        for (int e = 0; e < NEXP; ++e) { offsets[e] = acc; acc += counts[e]; cursor[e] = 0; }
        offsets[NEXP] = acc;
    }
}

__global__ void scatter_kernel(const int* __restrict__ texp, const float* __restrict__ twgt,
                               const int* __restrict__ offsets, int* __restrict__ cursor,
                               int* __restrict__ tok_id, float* __restrict__ tok_w) {
    int t = blockIdx.x * blockDim.x + threadIdx.x;
    if (t >= T_TOK) return;
    for (int j = 0; j < KTOP; ++j) {
        int e = texp[t * 2 + j];
        int pos = atomicAdd(&cursor[e], 1);
        int idx = offsets[e] + pos;
        tok_id[idx] = t;
        tok_w[idx]  = twgt[t * 2 + j];
    }
}

// ---------------- GEMM1: h = silu(x@w1^T) * (x@w3^T), bf16 WMMA ----------------
// block = 256 threads = 8 waves, wave grid 4(M) x 2(N); block tile 64 rows x 64 f-cols.
// k-loop double-buffered (ping/pong register sets); waves_per_eu(2,4) gives the
// allocator enough VGPRs to keep both buffers live so loads overlap the WMMA pipe.
__global__ __launch_bounds__(256)
__attribute__((amdgpu_waves_per_eu(2, 4)))
void gemm1_kernel(const bf16* __restrict__ xb, const bf16* __restrict__ w1b,
                  const bf16* __restrict__ w3b, const int* __restrict__ offsets,
                  const int* __restrict__ tok_id, bf16* __restrict__ hb) {
    const int e    = blockIdx.x;
    const int seg0 = offsets[e], seg1 = offsets[e + 1];
    const int m0   = seg0 + blockIdx.y * 64;
    if (m0 >= seg1) return;
    const int zb   = blockIdx.z * 64;

    const int wid  = threadIdx.x >> 5, lane = threadIdx.x & 31;
    const int wm   = (wid & 3) * 16;      // wave M offset within block
    const int wn   = (wid >> 2) * 32;     // wave N offset within block
    const int lrow = lane & 15, lhi = lane >> 4;

    int aidx = m0 + wm + lrow;
    if (aidx >= seg1) aidx = seg0;        // clamp to valid row (EXEC must stay full for WMMA)
    const bf16* aptr = xb + (size_t)tok_id[aidx] * DDIM;

    const int f0 = zb + wn + lrow;
    const bf16* b1p0 = w1b + ((size_t)e * FDIM + f0) * DDIM;
    const bf16* b1p1 = b1p0 + (size_t)16 * DDIM;
    const bf16* b3p0 = w3b + ((size_t)e * FDIM + f0) * DDIM;
    const bf16* b3p1 = b3p0 + (size_t)16 * DDIM;

    v8f g0 = {}, g1 = {}, u0 = {}, u1 = {};

    BFrag a0, p10, p11, p30, p31;   // ping set
    BFrag a1, q10, q11, q30, q31;   // pong set

    // prologue: ping set at k=0
    LOAD_AFRAG(a0,  aptr, 0, lhi);
    LOAD_BFRAG(p10, b1p0, 0, lhi);
    LOAD_BFRAG(p11, b1p1, 0, lhi);
    LOAD_BFRAG(p30, b3p0, 0, lhi);
    LOAD_BFRAG(p31, b3p1, 0, lhi);

    for (int k = 0; k < DDIM; k += 64) {
        // issue pong loads (k+32) before consuming ping
        LOAD_AFRAG(a1,  aptr, k + 32, lhi);
        LOAD_BFRAG(q10, b1p0, k + 32, lhi);
        LOAD_BFRAG(q11, b1p1, k + 32, lhi);
        LOAD_BFRAG(q30, b3p0, k + 32, lhi);
        LOAD_BFRAG(q31, b3p1, k + 32, lhi);

        WMMA_BF16(g0, a0, p10);
        WMMA_BF16(g1, a0, p11);
        WMMA_BF16(u0, a0, p30);
        WMMA_BF16(u1, a0, p31);

        if (k + 64 < DDIM) {    // uniform branch: prefetch ping for next iteration
            LOAD_AFRAG(a0,  aptr, k + 64, lhi);
            LOAD_BFRAG(p10, b1p0, k + 64, lhi);
            LOAD_BFRAG(p11, b1p1, k + 64, lhi);
            LOAD_BFRAG(p30, b3p0, k + 64, lhi);
            LOAD_BFRAG(p31, b3p1, k + 64, lhi);
        }

        WMMA_BF16(g0, a1, q10);
        WMMA_BF16(g1, a1, q11);
        WMMA_BF16(u0, a1, q30);
        WMMA_BF16(u1, a1, q31);
    }

    // C layout: VGPR j, lanes 0-15 -> M=j, lanes 16-31 -> M=j+8; N=lane%16
    for (int j = 0; j < 8; ++j) {
        int arow = m0 + wm + lhi * 8 + j;
        if (arow < seg1) {
            float g = g0[j], u = u0[j];
            float hv = (g / (1.0f + __expf(-g))) * u;           // silu(g) * u
            hb[(size_t)arow * FDIM + zb + wn + lrow] = to_bf16(hv);
            g = g1[j]; u = u1[j];
            hv = (g / (1.0f + __expf(-g))) * u;
            hb[(size_t)arow * FDIM + zb + wn + 16 + lrow] = to_bf16(hv);
        }
    }
}

// ---------------- GEMM2: out[t,:] += w * (h @ w2^T) ----------------
__global__ __launch_bounds__(256)
__attribute__((amdgpu_waves_per_eu(2, 4)))
void gemm2_kernel(const bf16* __restrict__ hb, const bf16* __restrict__ w2b,
                  const int* __restrict__ offsets, const int* __restrict__ tok_id,
                  const float* __restrict__ tok_w, float* __restrict__ out) {
    const int e    = blockIdx.x;
    const int seg0 = offsets[e], seg1 = offsets[e + 1];
    const int m0   = seg0 + blockIdx.y * 64;
    if (m0 >= seg1) return;
    const int zb   = blockIdx.z * 64;     // over D

    const int wid  = threadIdx.x >> 5, lane = threadIdx.x & 31;
    const int wm   = (wid & 3) * 16;
    const int wn   = (wid >> 2) * 32;
    const int lrow = lane & 15, lhi = lane >> 4;

    int aidx = m0 + wm + lrow;
    if (aidx >= seg1) aidx = seg0;
    const bf16* aptr = hb + (size_t)aidx * FDIM;

    const int d0 = zb + wn + lrow;
    const bf16* b2p0 = w2b + ((size_t)e * DDIM + d0) * FDIM;
    const bf16* b2p1 = b2p0 + (size_t)16 * FDIM;

    v8f c0 = {}, c1 = {};

    BFrag a0, p20, p21;   // ping
    BFrag a1, q20, q21;   // pong

    LOAD_AFRAG(a0,  aptr, 0, lhi);
    LOAD_BFRAG(p20, b2p0, 0, lhi);
    LOAD_BFRAG(p21, b2p1, 0, lhi);

    for (int k = 0; k < FDIM; k += 64) {
        LOAD_AFRAG(a1,  aptr, k + 32, lhi);
        LOAD_BFRAG(q20, b2p0, k + 32, lhi);
        LOAD_BFRAG(q21, b2p1, k + 32, lhi);

        WMMA_BF16(c0, a0, p20);
        WMMA_BF16(c1, a0, p21);

        if (k + 64 < FDIM) {
            LOAD_AFRAG(a0,  aptr, k + 64, lhi);
            LOAD_BFRAG(p20, b2p0, k + 64, lhi);
            LOAD_BFRAG(p21, b2p1, k + 64, lhi);
        }

        WMMA_BF16(c0, a1, q20);
        WMMA_BF16(c1, a1, q21);
    }

    for (int j = 0; j < 8; ++j) {
        int arow = m0 + wm + lhi * 8 + j;
        if (arow < seg1) {
            int   tok = tok_id[arow];
            float w   = tok_w[arow];
            atomicAdd(&out[(size_t)tok * DDIM + zb + wn + lrow],      w * c0[j]);
            atomicAdd(&out[(size_t)tok * DDIM + zb + wn + 16 + lrow], w * c1[j]);
        }
    }
}

// ---------------- host launcher ----------------
extern "C" void kernel_launch(void* const* d_in, const int* in_sizes, int n_in,
                              void* d_out, int out_size, void* d_ws, size_t ws_size,
                              hipStream_t stream) {
    const float* x  = (const float*)d_in[0];
    const float* w1 = (const float*)d_in[1];
    const float* w3 = (const float*)d_in[2];
    const float* w2 = (const float*)d_in[3];
    const float* rl = (const float*)d_in[4];
    // d_in[5] = top_k (fixed 2)
    float* out = (float*)d_out;

    char* ws = (char*)d_ws;
    size_t off = 0;
    auto alloc = [&](size_t bytes) -> void* {
        off = (off + 255) & ~(size_t)255;
        void* p = ws + off;
        off += bytes;
        return p;
    };
    const size_t WN = (size_t)NEXP * FDIM * DDIM;   // elements per weight tensor
    bf16* w1b = (bf16*)alloc(WN * 2);
    bf16* w3b = (bf16*)alloc(WN * 2);
    bf16* w2b = (bf16*)alloc(WN * 2);
    bf16* xb  = (bf16*)alloc((size_t)T_TOK * DDIM * 2);
    bf16* hb  = (bf16*)alloc((size_t)NASSIGN * FDIM * 2);
    int*   texp   = (int*)alloc((size_t)T_TOK * 2 * 4);
    float* twgt   = (float*)alloc((size_t)T_TOK * 2 * 4);
    int*   tok_id = (int*)alloc((size_t)NASSIGN * 4);
    float* tok_w  = (float*)alloc((size_t)NASSIGN * 4);
    int*   counts  = (int*)alloc(NEXP * 4);
    int*   offsets = (int*)alloc((NEXP + 1) * 4);
    int*   cursor  = (int*)alloc(NEXP * 4);

    // zero accumulators / counters (counts..cursor are one contiguous 256B-aligned span)
    hipMemsetAsync(d_out, 0, (size_t)T_TOK * DDIM * sizeof(float), stream);
    hipMemsetAsync(counts, 0, ((char*)(cursor + NEXP)) - (char*)counts, stream);

    // fp32 -> bf16 conversions
    {
        long long n4w = (long long)(WN / 4);
        long long n4x = (long long)((size_t)T_TOK * DDIM / 4);
        cvt_f32_bf16_kernel<<<4096, 256, 0, stream>>>(w1, (unsigned short*)w1b, n4w);
        cvt_f32_bf16_kernel<<<4096, 256, 0, stream>>>(w3, (unsigned short*)w3b, n4w);
        cvt_f32_bf16_kernel<<<4096, 256, 0, stream>>>(w2, (unsigned short*)w2b, n4w);
        cvt_f32_bf16_kernel<<<2048, 256, 0, stream>>>(x,  (unsigned short*)xb,  n4x);
    }

    // routing
    route_kernel<<<T_TOK / 256, 256, 0, stream>>>(rl, counts, texp, twgt);
    scan_kernel<<<1, 32, 0, stream>>>(counts, offsets, cursor);
    scatter_kernel<<<T_TOK / 256, 256, 0, stream>>>(texp, twgt, offsets, cursor, tok_id, tok_w);

    // expert GEMMs
    gemm1_kernel<<<dim3(NEXP, T_TOK / 64, FDIM / 64), 256, 0, stream>>>(
        xb, w1b, w3b, offsets, tok_id, hb);
    gemm2_kernel<<<dim3(NEXP, T_TOK / 64, DDIM / 64), 256, 0, stream>>>(
        hb, w2b, offsets, tok_id, tok_w, out);
}
